// Decoder_84172769067908
// MI455X (gfx1250) — matrix operand order repaired and checked by reading, not compiled, not run on hardware
//
#include <hip/hip_runtime.h>
#include <hip/hip_bf16.h>
#include <cstdint>

#define EMB   400
#define HID   300
#define KEY   128
#define VAL   128
#define VOCAB 33
#define TT    256
#define BB    256
#define SS    512
#define XK    832           // padded concat: 400(emb)+128(ctx)+300(h)+4(pad) = 13*64
#define G4    1200          // 4*HID
#define KSTAGE  64
#define NSTAGES (XK / KSTAGE)   // 13

typedef __attribute__((ext_vector_type(16))) __bf16    v16bf;
typedef __attribute__((ext_vector_type(8)))  float     v8f;
typedef __attribute__((ext_vector_type(4)))  unsigned  v4u;

union BFrag { v16bf v; v4u q[2]; };

// ---------------------------------------------------------------- utilities
__global__ void f32_to_bf16_kernel(const float* __restrict__ src,
                                   __hip_bfloat16* __restrict__ dst, int n) {
    int i = blockIdx.x * blockDim.x + threadIdx.x;
    int stride = gridDim.x * blockDim.x;
    for (; i < n; i += stride) dst[i] = __float2bfloat16(src[i]);
}

// Wcat[n,k]: k<528 -> W_ih[n,k]; 528<=k<828 -> W_hh[n,k-528]; else 0
__global__ void build_wcat_kernel(const float* __restrict__ W_ih,
                                  const float* __restrict__ W_hh,
                                  __hip_bfloat16* __restrict__ Wcat) {
    int idx = blockIdx.x * blockDim.x + threadIdx.x;
    if (idx >= G4 * XK) return;
    int n = idx / XK, k = idx % XK;
    float v = 0.f;
    if (k < EMB + VAL)            v = W_ih[n * (EMB + VAL) + k];
    else if (k < EMB + VAL + HID) v = W_hh[n * HID + (k - EMB - VAL)];
    Wcat[idx] = __float2bfloat16(v);
}

// ---------------------------------------------------------------- attention
__device__ __forceinline__ void attend_block(
    int b, int tid, const float* sh, float* sq, float* se, float* sred,
    float* sctx2, const float* __restrict__ phi_w, const float* __restrict__ phi_b,
    const __hip_bfloat16* __restrict__ keysb,
    const __hip_bfloat16* __restrict__ valsb, float* __restrict__ ctx_g) {
    if (tid < KEY) {
        const float* pw = phi_w + (size_t)tid * HID;
        float q = phi_b[tid];
        for (int k = 0; k < HID; ++k) q += sh[k] * pw[k];
        sq[tid] = q;
    }
    __syncthreads();

    for (int s = tid; s < SS; s += 256) {
        const __hip_bfloat16* kp = keysb + ((size_t)s * BB + b) * KEY;
        __builtin_prefetch((const void*)(valsb + ((size_t)s * BB + b) * VAL), 0, 1);
        float e = 0.f;
        for (int k = 0; k < KEY; ++k) e += sq[k] * __bfloat162float(kp[k]);
        se[s] = e;
    }
    __syncthreads();

    sred[tid] = fmaxf(se[tid], se[tid + 256]);
    __syncthreads();
    for (int off = 128; off > 0; off >>= 1) {
        if (tid < off) sred[tid] = fmaxf(sred[tid], sred[tid + off]);
        __syncthreads();
    }
    float mx = sred[0];
    __syncthreads();

    float partial = 0.f;
    for (int s = tid; s < SS; s += 256) {
        float e = __expf(se[s] - mx);
        se[s] = e;
        partial += e;
    }
    sred[tid] = partial;
    __syncthreads();
    for (int off = 128; off > 0; off >>= 1) {
        if (tid < off) sred[tid] += sred[tid + off];
        __syncthreads();
    }
    float inv = 1.f / sred[0];
    __syncthreads();
    for (int s = tid; s < SS; s += 256) se[s] *= inv;
    __syncthreads();

    if (tid < VAL) {
        const __hip_bfloat16* vp = valsb + (size_t)b * VAL + tid;
        float acc = 0.f;
        for (int s = 0; s < SS; ++s)
            acc += se[s] * __bfloat162float(vp[(size_t)s * BB * VAL]);
        sctx2[tid] = acc;
        ctx_g[b * VAL + tid] = acc;
    }
}

// ------------------------------------------------ gates GEMM via bf16 WMMA
// G[256,1200] = X[256,832](bf16) @ Wcat[1200,832]^T(bf16), f32 accum.
// 30 blocks x 8 waves. Block: 128 rows x 80 cols; wave: 16 rows x 80 cols.
// 64-wide K stages double-buffered in LDS via straight-line async copies.
// Per 32-wide K-step: preload A + all 5 B fragments (12 clustered
// ds_load_b128), then 5 back-to-back WMMAs with no interleaved waits.
__global__ __launch_bounds__(256)
void gates_wmma_kernel(const __hip_bfloat16* __restrict__ X,
                       const __hip_bfloat16* __restrict__ W,
                       float* __restrict__ G) {
    __shared__ __align__(16) __hip_bfloat16 sA[2][128 * KSTAGE];   // 16 KB each
    __shared__ __align__(16) __hip_bfloat16 sW[2][80 * KSTAGE];    // 10 KB each

    const int tid  = threadIdx.x;
    const int mblk = blockIdx.x / 15, nblk = blockIdx.x % 15;
    const int m0   = mblk * 128,      n0   = nblk * 80;
    const int wave = tid >> 5, lane = tid & 31;
    const int grp  = lane >> 4, lr  = lane & 15;

    // 16B chunk ci of a stage: row = ci>>3, k-offset = (ci&7)*8 elements;
    // LDS destination is base + ci*16 (row-major, stride 64 bf16).
    auto issue_stage = [&](int buf, int k0) {
        uint32_t abase = (uint32_t)(uintptr_t)(&sA[buf][0]) + (uint32_t)tid * 16u;
        const __hip_bfloat16* Xb = X + (size_t)m0 * XK + k0;
#pragma unroll
        for (int u = 0; u < 4; ++u) {            // 1024 A chunks / 256 thr
            int ci = tid + u * 256;
            int row = ci >> 3, kc = (ci & 7) * 8;
            uint64_t g = (uint64_t)(uintptr_t)(Xb + (size_t)row * XK + kc);
            uint32_t l = abase + (uint32_t)u * 4096u;
            asm volatile("global_load_async_to_lds_b128 %0, %1, off"
                         :: "v"(l), "v"(g) : "memory");
        }
        uint32_t wbase = (uint32_t)(uintptr_t)(&sW[buf][0]) + (uint32_t)tid * 16u;
        const __hip_bfloat16* Wb = W + (size_t)n0 * XK + k0;
#pragma unroll
        for (int u = 0; u < 2; ++u) {            // 640 W chunks: 2 each ...
            int ci = tid + u * 256;
            int row = ci >> 3, kc = (ci & 7) * 8;
            uint64_t g = (uint64_t)(uintptr_t)(Wb + (size_t)row * XK + kc);
            uint32_t l = wbase + (uint32_t)u * 4096u;
            asm volatile("global_load_async_to_lds_b128 %0, %1, off"
                         :: "v"(l), "v"(g) : "memory");
        }
        if (tid < 128) {                         // ... + predicated tail
            int ci = tid + 512;
            int row = ci >> 3, kc = (ci & 7) * 8;
            uint64_t g = (uint64_t)(uintptr_t)(Wb + (size_t)row * XK + kc);
            uint32_t l = wbase + 8192u;
            asm volatile("global_load_async_to_lds_b128 %0, %1, off"
                         :: "v"(l), "v"(g) : "memory");
        }
    };

    v8f acc[5] = {};

    issue_stage(0, 0);
    asm volatile("s_wait_asynccnt 0" ::: "memory");
    __syncthreads();

    for (int st = 0; st < NSTAGES; ++st) {
        int buf = st & 1;
        if (st + 1 < NSTAGES) issue_stage(buf ^ 1, (st + 1) * KSTAGE);

#pragma unroll
        for (int ks2 = 0; ks2 < 2; ++ks2) {      // two 32-wide WMMA K-steps
            // Preload A fragment + all 5 B fragments (clustered ds loads)
            BFrag a, bfr[5];
            const __hip_bfloat16* arow =
                &sA[buf][(wave * 16 + lr) * KSTAGE + ks2 * 32];
            a.q[0] = *(const v4u*)(arow + grp * 8);
            a.q[1] = *(const v4u*)(arow + 16 + grp * 8);
#pragma unroll
            for (int i = 0; i < 5; ++i) {
                const __hip_bfloat16* brow =
                    &sW[buf][(i * 16 + lr) * KSTAGE + ks2 * 32];
                bfr[i].q[0] = *(const v4u*)(brow + grp * 8);
                bfr[i].q[1] = *(const v4u*)(brow + 16 + grp * 8);
            }
            // Back-to-back WMMA chain (independent accumulators)
#pragma unroll
            for (int i = 0; i < 5; ++i)
                acc[i] = __builtin_amdgcn_wmma_f32_16x16x32_bf16(
                    false, a.v, false, bfr[i].v, (short)0, acc[i], false, false);
        }

        asm volatile("s_wait_asynccnt 0" ::: "memory");
        __syncthreads();
    }

#pragma unroll
    for (int i = 0; i < 5; ++i)
#pragma unroll
        for (int r = 0; r < 8; ++r)
            G[(size_t)(m0 + wave * 16 + grp * 8 + r) * G4 + n0 + i * 16 + lr] =
                acc[i][r];
}

// ------------------------------------------------- fused per-step cell
__global__ __launch_bounds__(256)
void fused_cell_kernel(const float* __restrict__ G,
                       const float* __restrict__ b_ih,
                       const float* __restrict__ b_hh,
                       float* __restrict__ c, float* __restrict__ ctx,
                       const float* __restrict__ proj_w,
                       const float* __restrict__ proj_b,
                       const float* __restrict__ phi_w,
                       const float* __restrict__ phi_b,
                       const __hip_bfloat16* __restrict__ keysb,
                       const __hip_bfloat16* __restrict__ valsb,
                       const int* __restrict__ input,
                       const float* __restrict__ embedding,
                       __hip_bfloat16* __restrict__ xcat,
                       float* __restrict__ out, int t) {
    __shared__ float sh[HID];      // h_new
    __shared__ float sctx[VAL];    // ctx (old, for logits)
    __shared__ float sctx2[VAL];   // ctx (new)
    __shared__ float sq[KEY];
    __shared__ float se[SS];
    __shared__ float sred[256];
    int b = blockIdx.x, tid = threadIdx.x;

    // LSTM cell
    for (int j = tid; j < HID; j += 256) {
        const float* g = G + (size_t)b * G4;
        float iv = g[j]           + b_ih[j]           + b_hh[j];
        float fv = g[HID + j]     + b_ih[HID + j]     + b_hh[HID + j];
        float gv = g[2 * HID + j] + b_ih[2 * HID + j] + b_hh[2 * HID + j];
        float ov = g[3 * HID + j] + b_ih[3 * HID + j] + b_hh[3 * HID + j];
        float si = 1.f / (1.f + __expf(-iv));
        float sf = 1.f / (1.f + __expf(-fv));
        float so = 1.f / (1.f + __expf(-ov));
        float cn = sf * c[b * HID + j] + si * tanhf(gv);
        c[b * HID + j] = cn;
        sh[j] = so * tanhf(cn);
    }
    for (int j = tid; j < VAL; j += 256) sctx[j] = ctx[b * VAL + j];
    __syncthreads();

    // logits(t) from (h_new, ctx_old)
    if (tid < VOCAB) {
        const float* pw = proj_w + (size_t)tid * (HID + VAL);
        float acc = proj_b[tid];
        for (int j = 0; j < HID; ++j) acc += sh[j] * pw[j];
        for (int d = 0; d < VAL; ++d) acc += sctx[d] * pw[HID + d];
        out[((size_t)t * BB + b) * VOCAB + tid] = acc;
    }
    __syncthreads();

    // ctx_new = attend(h_new)
    attend_block(b, tid, sh, sq, se, sred, sctx2, phi_w, phi_b, keysb, valsb, ctx);
    __syncthreads();

    // build next step's X row: [emb(x_{t+1}) | ctx_new | h_new | 0pad] bf16
    if (t + 1 < TT) {
        int tok = input[(t + 1) * BB + b];
        for (int k = tid; k < XK; k += 256) {
            float v;
            if (k < EMB)                  v = embedding[tok * EMB + k];
            else if (k < EMB + VAL)       v = sctx2[k - EMB];
            else if (k < EMB + VAL + HID) v = sh[k - EMB - VAL];
            else                          v = 0.f;
            xcat[(size_t)b * XK + k] = __float2bfloat16(v);
        }
    }
}

// -------------------------------------------------------------- prolog
__global__ __launch_bounds__(256)
void prolog_kernel(const float* __restrict__ h0, const float* __restrict__ c0,
                   float* __restrict__ c, float* __restrict__ ctx,
                   const float* __restrict__ phi_w,
                   const float* __restrict__ phi_b,
                   const __hip_bfloat16* __restrict__ keysb,
                   const __hip_bfloat16* __restrict__ valsb,
                   const int* __restrict__ input,
                   const float* __restrict__ embedding,
                   __hip_bfloat16* __restrict__ xcat) {
    __shared__ float sh[HID];
    __shared__ float sctx2[VAL];
    __shared__ float sq[KEY];
    __shared__ float se[SS];
    __shared__ float sred[256];
    int b = blockIdx.x, tid = threadIdx.x;

    for (int j = tid; j < HID; j += 256) {
        sh[j] = h0[j];
        c[b * HID + j] = c0[j];
    }
    __syncthreads();

    attend_block(b, tid, sh, sq, se, sred, sctx2, phi_w, phi_b, keysb, valsb, ctx);
    __syncthreads();

    int tok = input[b];  // t = 0
    for (int k = tid; k < XK; k += 256) {
        float v;
        if (k < EMB)                  v = embedding[tok * EMB + k];
        else if (k < EMB + VAL)       v = sctx2[k - EMB];
        else if (k < EMB + VAL + HID) v = sh[k - EMB - VAL];
        else                          v = 0.f;
        xcat[(size_t)b * XK + k] = __float2bfloat16(v);
    }
}

// ------------------------------------------------------------------ launch
extern "C" void kernel_launch(void* const* d_in, const int* in_sizes, int n_in,
                              void* d_out, int out_size, void* d_ws, size_t ws_size,
                              hipStream_t stream) {
    const int*   input     = (const int*)  d_in[0];
    const float* keys      = (const float*)d_in[1];
    const float* values    = (const float*)d_in[2];
    const float* embedding = (const float*)d_in[3];
    const float* phi_w     = (const float*)d_in[4];
    const float* phi_b     = (const float*)d_in[5];
    const float* h0        = (const float*)d_in[6];
    const float* c0        = (const float*)d_in[7];
    const float* W_ih      = (const float*)d_in[8];
    const float* b_ih      = (const float*)d_in[9];
    const float* W_hh      = (const float*)d_in[10];
    const float* b_hh      = (const float*)d_in[11];
    const float* proj_w    = (const float*)d_in[12];
    const float* proj_b    = (const float*)d_in[13];
    float* out = (float*)d_out;

    char* p = (char*)d_ws;
    const size_t KV = (size_t)SS * BB * KEY;              // 16,777,216 elems
    __hip_bfloat16* keysb = (__hip_bfloat16*)p;  p += KV * 2;
    __hip_bfloat16* valsb = (__hip_bfloat16*)p;  p += KV * 2;
    __hip_bfloat16* Wcat  = (__hip_bfloat16*)p;  p += (size_t)G4 * XK * 2;
    __hip_bfloat16* xcat  = (__hip_bfloat16*)p;  p += (size_t)BB * XK * 2;
    float* G   = (float*)p;  p += (size_t)BB * G4 * 4;
    float* c   = (float*)p;  p += (size_t)BB * HID * 4;
    float* ctx = (float*)p;  p += (size_t)BB * VAL * 4;

    f32_to_bf16_kernel<<<4096, 256, 0, stream>>>(keys,   keysb, (int)KV);
    f32_to_bf16_kernel<<<4096, 256, 0, stream>>>(values, valsb, (int)KV);
    build_wcat_kernel<<<(G4 * XK + 255) / 256, 256, 0, stream>>>(W_ih, W_hh, Wcat);
    prolog_kernel<<<BB, 256, 0, stream>>>(h0, c0, c, ctx, phi_w, phi_b,
                                          keysb, valsb, input, embedding, xcat);

    for (int t = 0; t < TT; ++t) {
        gates_wmma_kernel<<<30, 256, 0, stream>>>(xcat, Wcat, G);
        fused_cell_kernel<<<BB, 256, 0, stream>>>(G, b_ih, b_hh, c, ctx,
                                                  proj_w, proj_b, phi_w, phi_b,
                                                  keysb, valsb, input, embedding,
                                                  xcat, out, t);
    }
}
